// LearnedSimulator_33449205301608
// MI455X (gfx1250) — matrix operand order repaired and checked by reading, not compiled, not run on hardware
//
#include <hip/hip_runtime.h>
#include <hip/hip_bf16.h>

#define NN 131072
#define NE 524288
#define H  128

typedef __attribute__((ext_vector_type(16))) _Float16 v16h;
typedef __attribute__((ext_vector_type(8)))  _Float16 v8h;
typedef __attribute__((ext_vector_type(8)))  float    v8f;
typedef __attribute__((ext_vector_type(4)))  float    v4f;

// ---------------------------------------------------------------------------
// WMMA helpers (gfx1250: D = A(16x32 f16) x B(32x16 f16) + C(16x16 f32))
// ---------------------------------------------------------------------------
__device__ __forceinline__ v8f wmma_f16(v16h a, v16h b, v8f c) {
  return __builtin_amdgcn_wmma_f32_16x16x32_f16(false, a, false, b, (short)0, c,
                                                false, false);
}

// A-fragment from two 8-halfs chunks: element e -> K = kbase + (e<8 ? e : 8+e)
__device__ __forceinline__ v16h afrag(v8h lo, v8h hi) {
  v16h a;
#pragma unroll
  for (int i = 0; i < 8; ++i) { a[i] = lo[i]; a[8 + i] = hi[i]; }
  return a;
}

__device__ __forceinline__ void zacc(v8f acc[2][8]) {
#pragma unroll
  for (int mt = 0; mt < 2; ++mt)
#pragma unroll
    for (int t = 0; t < 8; ++t)
#pragma unroll
      for (int v = 0; v < 8; ++v) acc[mt][t][v] = 0.f;
}

// K=128 GEMM layer reading A from per-wave LDS buffer (2 tiles x 16 rows x 128)
__device__ __forceinline__ void layer_lds(const _Float16* buf,
                                          const _Float16* __restrict__ wp,
                                          v8f acc[2][8], int lane) {
  const int m = lane & 15, kb = (lane >> 4) * 8;
#pragma unroll
  for (int s = 0; s < 4; ++s) {
    v16h a[2];
#pragma unroll
    for (int mt = 0; mt < 2; ++mt) {
      const _Float16* rp = buf + (mt * 16 + m) * H + s * 32 + kb;
      a[mt] = afrag(*(const v8h*)rp, *(const v8h*)(rp + 16));
    }
#pragma unroll
    for (int t = 0; t < 8; ++t) {
      v16h b = *(const v16h*)(wp + ((size_t)(s * 8 + t) * 32 + lane) * 16);
#pragma unroll
      for (int mt = 0; mt < 2; ++mt) acc[mt][t] = wmma_f16(a[mt], b, acc[mt][t]);
    }
  }
}

// bias + relu, C-layout -> row-major f16 LDS buffer
__device__ __forceinline__ void bias_relu_to_lds(v8f acc[2][8],
                                                 const float* __restrict__ bias,
                                                 _Float16* buf, int lane) {
  const int n15 = lane & 15, hb = lane >> 4;
#pragma unroll
  for (int t = 0; t < 8; ++t) {
    float bv = bias[t * 16 + n15];
#pragma unroll
    for (int mt = 0; mt < 2; ++mt)
#pragma unroll
      for (int v = 0; v < 8; ++v) {
        float x = acc[mt][t][v] + bv;
        buf[(mt * 16 + hb * 8 + v) * H + t * 16 + n15] = (_Float16)fmaxf(x, 0.f);
      }
  }
}

__device__ __forceinline__ void store_to_lds(v8f acc[2][8], _Float16* buf,
                                             int lane) {
  const int n15 = lane & 15, hb = lane >> 4;
#pragma unroll
  for (int t = 0; t < 8; ++t)
#pragma unroll
    for (int mt = 0; mt < 2; ++mt)
#pragma unroll
      for (int v = 0; v < 8; ++v)
        buf[(mt * 16 + hb * 8 + v) * H + t * 16 + n15] = (_Float16)acc[mt][t][v];
}

// bias + LayerNorm over 128 features; rows live across 16-lane halves
__device__ __forceinline__ void bias_ln(v8f acc[2][8],
                                        const float* __restrict__ bias,
                                        const float* __restrict__ g,
                                        const float* __restrict__ beta,
                                        int lane) {
  const int n15 = lane & 15;
  float gv[8], bt[8];
#pragma unroll
  for (int t = 0; t < 8; ++t) {
    float bv = bias[t * 16 + n15];
    gv[t] = g[t * 16 + n15];
    bt[t] = beta[t * 16 + n15];
#pragma unroll
    for (int mt = 0; mt < 2; ++mt)
#pragma unroll
      for (int v = 0; v < 8; ++v) acc[mt][t][v] += bv;
  }
#pragma unroll
  for (int mt = 0; mt < 2; ++mt)
#pragma unroll
    for (int v = 0; v < 8; ++v) {
      float s = 0.f, ss = 0.f;
#pragma unroll
      for (int t = 0; t < 8; ++t) { float x = acc[mt][t][v]; s += x; ss += x * x; }
#pragma unroll
      for (int msk = 1; msk < 16; msk <<= 1) {
        s  += __shfl_xor(s,  msk, 32);
        ss += __shfl_xor(ss, msk, 32);
      }
      float mu  = s  * (1.f / 128.f);
      float var = ss * (1.f / 128.f) - mu * mu;
      float rs  = rsqrtf(var + 1e-5f);
#pragma unroll
      for (int t = 0; t < 8; ++t)
        acc[mt][t][v] = (acc[mt][t][v] - mu) * rs * gv[t] + bt[t];
    }
}

// ---------------------------------------------------------------------------
// Encoder MLP: in (rows x 32 f16, zero padded) -> 128 -> 128 -> 128 + LN
// ---------------------------------------------------------------------------
__global__ __launch_bounds__(256) void mlp_enc_kernel(
    const _Float16* __restrict__ in,
    const _Float16* __restrict__ w0p, const _Float16* __restrict__ w1p,
    const _Float16* __restrict__ w2p,
    const float* __restrict__ b0, const float* __restrict__ b1,
    const float* __restrict__ b2,
    const float* __restrict__ g, const float* __restrict__ beta,
    _Float16* __restrict__ out) {
  __shared__ _Float16 lbuf[32768];
  const int lane = threadIdx.x & 31, wid = threadIdx.x >> 5;
  const int tb = (blockIdx.x * 8 + wid) * 32;
  _Float16* buf = lbuf + wid * 4096;
  const int m = lane & 15, kb = (lane >> 4) * 8;

  v8f acc[2][8];
  zacc(acc);
  {  // layer 1: K = 32 (single k-step)
    v16h a[2];
#pragma unroll
    for (int mt = 0; mt < 2; ++mt) {
      const _Float16* rp = in + (size_t)(tb + mt * 16 + m) * 32 + kb;
      a[mt] = afrag(*(const v8h*)rp, *(const v8h*)(rp + 16));
    }
#pragma unroll
    for (int t = 0; t < 8; ++t) {
      v16h b = *(const v16h*)(w0p + ((size_t)t * 32 + lane) * 16);
#pragma unroll
      for (int mt = 0; mt < 2; ++mt) acc[mt][t] = wmma_f16(a[mt], b, acc[mt][t]);
    }
  }
  bias_relu_to_lds(acc, b0, buf, lane);
  zacc(acc); layer_lds(buf, w1p, acc, lane); bias_relu_to_lds(acc, b1, buf, lane);
  zacc(acc); layer_lds(buf, w2p, acc, lane); bias_ln(acc, b2, g, beta, lane);
  store_to_lds(acc, buf, lane);
  for (int idx = lane; idx < 512; idx += 32) {
    int row = idx >> 4, c8 = (idx & 15) * 8;
    *(v8h*)(out + (size_t)(tb + row) * H + c8) = *(const v8h*)(buf + row * H + c8);
  }
}

// ---------------------------------------------------------------------------
// Edge message MLP: concat(h[i],h[j],e)(384) -> 128 -> 128 -> 128 + LN,
// fused residual e += m and f32 atomic scatter-add into aggr[dst]
// ---------------------------------------------------------------------------
__global__ __launch_bounds__(256) void edge_mp_kernel(
    const _Float16* __restrict__ h, _Float16* __restrict__ e,
    const int* __restrict__ effi, const int* __restrict__ effj,
    const int* __restrict__ dst,
    const _Float16* __restrict__ w0p, const _Float16* __restrict__ w1p,
    const _Float16* __restrict__ w2p,
    const float* __restrict__ b0, const float* __restrict__ b1,
    const float* __restrict__ b2,
    const float* __restrict__ g, const float* __restrict__ beta,
    float* __restrict__ aggr) {
  __shared__ _Float16 lbuf[32768];
  const int lane = threadIdx.x & 31, wid = threadIdx.x >> 5;
  const int tb = (blockIdx.x * 8 + wid) * 32;
  _Float16* buf = lbuf + wid * 4096;
  const int m = lane & 15, kb = (lane >> 4) * 8;

  const _Float16* pA[2][3];
#pragma unroll
  for (int mt = 0; mt < 2; ++mt) {
    int r = tb + mt * 16 + m;
    pA[mt][0] = h + (size_t)effi[r] * H;
    pA[mt][1] = h + (size_t)effj[r] * H;
    pA[mt][2] = e + (size_t)r * H;
  }

  v8f acc[2][8];
  zacc(acc);
#pragma unroll
  for (int s = 0; s < 12; ++s) {  // K = 384
    const int seg = s >> 2, kk = (s & 3) * 32 + kb;
    v16h a[2];
#pragma unroll
    for (int mt = 0; mt < 2; ++mt) {
      const _Float16* p = pA[mt][seg] + kk;
      a[mt] = afrag(*(const v8h*)p, *(const v8h*)(p + 16));
    }
#pragma unroll
    for (int t = 0; t < 8; ++t) {
      v16h b = *(const v16h*)(w0p + ((size_t)(s * 8 + t) * 32 + lane) * 16);
#pragma unroll
      for (int mt = 0; mt < 2; ++mt) acc[mt][t] = wmma_f16(a[mt], b, acc[mt][t]);
    }
  }
  bias_relu_to_lds(acc, b0, buf, lane);
  zacc(acc); layer_lds(buf, w1p, acc, lane); bias_relu_to_lds(acc, b1, buf, lane);
  zacc(acc); layer_lds(buf, w2p, acc, lane); bias_ln(acc, b2, g, beta, lane);
  store_to_lds(acc, buf, lane);

  for (int idx = lane; idx < 512; idx += 32) {
    int row = idx >> 4, c8 = (idx & 15) * 8;
    int r = tb + row;
    v8h mv = *(const v8h*)(buf + row * H + c8);
    _Float16* ep = e + (size_t)r * H + c8;
    v8h ev = *(const v8h*)ep;
#pragma unroll
    for (int i = 0; i < 8; ++i) ev[i] = (_Float16)((float)ev[i] + (float)mv[i]);
    *(v8h*)ep = ev;
    float* ap = aggr + (size_t)dst[r] * H + c8;
#pragma unroll
    for (int i = 0; i < 8; ++i)
      __hip_atomic_fetch_add(ap + i, (float)mv[i], __ATOMIC_RELAXED,
                             __HIP_MEMORY_SCOPE_AGENT);
  }
}

// ---------------------------------------------------------------------------
// Node update MLP: concat(h, aggr)(256) -> 128 -> 128 -> 128 + LN; h += out
// ---------------------------------------------------------------------------
__global__ __launch_bounds__(256) void node_mp_kernel(
    _Float16* __restrict__ h, const float* __restrict__ aggr,
    const _Float16* __restrict__ w0p, const _Float16* __restrict__ w1p,
    const _Float16* __restrict__ w2p,
    const float* __restrict__ b0, const float* __restrict__ b1,
    const float* __restrict__ b2,
    const float* __restrict__ g, const float* __restrict__ beta) {
  __shared__ _Float16 lbuf[32768];
  const int lane = threadIdx.x & 31, wid = threadIdx.x >> 5;
  const int tb = (blockIdx.x * 8 + wid) * 32;
  _Float16* buf = lbuf + wid * 4096;
  const int m = lane & 15, kb = (lane >> 4) * 8;

  v8f acc[2][8];
  zacc(acc);
#pragma unroll
  for (int s = 0; s < 8; ++s) {  // K = 256 (h f16 | aggr f32)
    v16h a[2];
#pragma unroll
    for (int mt = 0; mt < 2; ++mt) {
      int r = tb + mt * 16 + m;
      if (s < 4) {
        const _Float16* p = h + (size_t)r * H + s * 32 + kb;
        a[mt] = afrag(*(const v8h*)p, *(const v8h*)(p + 16));
      } else {
        const float* ap = aggr + (size_t)r * H + (s - 4) * 32 + kb;
        v8h lo, hi;
#pragma unroll
        for (int i = 0; i < 8; ++i) {
          lo[i] = (_Float16)ap[i];
          hi[i] = (_Float16)ap[16 + i];
        }
        a[mt] = afrag(lo, hi);
      }
    }
#pragma unroll
    for (int t = 0; t < 8; ++t) {
      v16h b = *(const v16h*)(w0p + ((size_t)(s * 8 + t) * 32 + lane) * 16);
#pragma unroll
      for (int mt = 0; mt < 2; ++mt) acc[mt][t] = wmma_f16(a[mt], b, acc[mt][t]);
    }
  }
  bias_relu_to_lds(acc, b0, buf, lane);
  zacc(acc); layer_lds(buf, w1p, acc, lane); bias_relu_to_lds(acc, b1, buf, lane);
  zacc(acc); layer_lds(buf, w2p, acc, lane); bias_ln(acc, b2, g, beta, lane);
  store_to_lds(acc, buf, lane);

  for (int idx = lane; idx < 512; idx += 32) {  // h += out
    int row = idx >> 4, c8 = (idx & 15) * 8;
    _Float16* hp = h + (size_t)(tb + row) * H + c8;
    v8h ov = *(const v8h*)(buf + row * H + c8);
    v8h hv = *(const v8h*)hp;
#pragma unroll
    for (int i = 0; i < 8; ++i) hv[i] = (_Float16)((float)hv[i] + (float)ov[i]);
    *(v8h*)hp = hv;
  }
}

// ---------------------------------------------------------------------------
// Decoder: h -> 128 -> 128 -> 1 (no LN); final layer as shuffled dot-product
// ---------------------------------------------------------------------------
__global__ __launch_bounds__(256) void decode_kernel(
    const _Float16* __restrict__ h,
    const _Float16* __restrict__ w0p, const _Float16* __restrict__ w1p,
    const float* __restrict__ b0, const float* __restrict__ b1,
    const float* __restrict__ w2, const float* __restrict__ b2,
    float* __restrict__ out) {
  __shared__ _Float16 lbuf[32768];
  const int lane = threadIdx.x & 31, wid = threadIdx.x >> 5;
  const int tb = (blockIdx.x * 8 + wid) * 32;
  _Float16* buf = lbuf + wid * 4096;
  const int m = lane & 15, kb = (lane >> 4) * 8;
  const int n15 = lane & 15, hb = lane >> 4;

  v8f acc[2][8];
  zacc(acc);
#pragma unroll
  for (int s = 0; s < 4; ++s) {
    v16h a[2];
#pragma unroll
    for (int mt = 0; mt < 2; ++mt) {
      const _Float16* p = h + (size_t)(tb + mt * 16 + m) * H + s * 32 + kb;
      a[mt] = afrag(*(const v8h*)p, *(const v8h*)(p + 16));
    }
#pragma unroll
    for (int t = 0; t < 8; ++t) {
      v16h b = *(const v16h*)(w0p + ((size_t)(s * 8 + t) * 32 + lane) * 16);
#pragma unroll
      for (int mt = 0; mt < 2; ++mt) acc[mt][t] = wmma_f16(a[mt], b, acc[mt][t]);
    }
  }
  bias_relu_to_lds(acc, b0, buf, lane);
  zacc(acc); layer_lds(buf, w1p, acc, lane);
#pragma unroll
  for (int t = 0; t < 8; ++t) {  // bias + relu in-register
    float bv = b1[t * 16 + n15];
#pragma unroll
    for (int mt = 0; mt < 2; ++mt)
#pragma unroll
      for (int v = 0; v < 8; ++v)
        acc[mt][t][v] = fmaxf(acc[mt][t][v] + bv, 0.f);
  }
  float w2v[8];
#pragma unroll
  for (int t = 0; t < 8; ++t) w2v[t] = w2[t * 16 + n15];
  float b2v = b2[0];
#pragma unroll
  for (int mt = 0; mt < 2; ++mt)
#pragma unroll
    for (int v = 0; v < 8; ++v) {
      float d = 0.f;
#pragma unroll
      for (int t = 0; t < 8; ++t) d += acc[mt][t][v] * w2v[t];
#pragma unroll
      for (int msk = 1; msk < 16; msk <<= 1) d += __shfl_xor(d, msk, 32);
      if (n15 == 0) out[tb + mt * 16 + hb * 8 + v] = d + b2v;
    }
}

// ---------------------------------------------------------------------------
// Small utility kernels
// ---------------------------------------------------------------------------
__global__ void pack_w_kernel(const float* __restrict__ W,
                              _Float16* __restrict__ P, int kin, int total) {
  int i = blockIdx.x * 256 + threadIdx.x;
  if (i >= total) return;
  int e = i & 15, lane = (i >> 4) & 31, t = (i >> 9) & 7, s = i >> 12;
  int n = t * 16 + (lane & 15);
  int kB = (lane < 16) ? 0 : 8;
  int K = s * 32 + kB + (e < 8 ? e : e + 8);
  P[i] = (_Float16)((K < kin) ? W[(size_t)K * 128 + n] : 0.f);
}

__global__ void build_nf_kernel(const float* __restrict__ x,
                                const int* __restrict__ y_type,
                                const float* __restrict__ y_rest,
                                const float* __restrict__ nerr,
                                const float* __restrict__ embed,
                                _Float16* __restrict__ nf) {
  int r = blockIdx.x * 256 + threadIdx.x;
  if (r >= NN) return;
  const float* em = embed + y_type[r] * 8;
  _Float16* o = nf + (size_t)r * 32;
#pragma unroll
  for (int i = 0; i < 8; ++i) o[i] = (_Float16)em[i];
  o[8]  = (_Float16)y_rest[r];
  o[9]  = (_Float16)x[r * 2];
  o[10] = (_Float16)x[r * 2 + 1];
  o[11] = (_Float16)nerr[r];
#pragma unroll
  for (int i = 12; i < 32; ++i) o[i] = (_Float16)0.f;
}

__global__ void build_ea_kernel(const float* __restrict__ ea,
                                _Float16* __restrict__ o16) {
  int r = blockIdx.x * 256 + threadIdx.x;
  if (r >= NE) return;
  _Float16* o = o16 + (size_t)r * 32;
#pragma unroll
  for (int i = 0; i < 5; ++i) o[i] = (_Float16)ea[(size_t)r * 5 + i];
#pragma unroll
  for (int i = 5; i < 32; ++i) o[i] = (_Float16)0.f;
}

__global__ void eff_idx_kernel(const int* __restrict__ ei,
                               const int* __restrict__ dn,
                               int* __restrict__ effi, int* __restrict__ effj) {
  int r = blockIdx.x * 256 + threadIdx.x;
  if (r >= NE) return;
  int d0 = dn[(size_t)r * 2], d1 = dn[(size_t)r * 2 + 1];
  if (d0 >= 0) {
    effi[r] = min(d0, NN - 1);
    effj[r] = min(max(d1, 0), NN - 1);
  } else {
    effi[r] = ei[NE + r];  // dst
    effj[r] = ei[r];       // src
  }
}

__global__ void zero_aggr_kernel(float* __restrict__ a) {
  size_t i = ((size_t)blockIdx.x * 256 + threadIdx.x) * 4;
  if (i < (size_t)NN * H) *(v4f*)(a + i) = v4f{0.f, 0.f, 0.f, 0.f};
}

__global__ void copy_err_kernel(const float* __restrict__ nerr,
                                float* __restrict__ out) {
  int r = blockIdx.x * 256 + threadIdx.x;
  if (r < NN) out[r] = nerr[r];
}

// ---------------------------------------------------------------------------
// Host launch
// ---------------------------------------------------------------------------
extern "C" void kernel_launch(void* const* d_in, const int* in_sizes, int n_in,
                              void* d_out, int out_size, void* d_ws,
                              size_t ws_size, hipStream_t stream) {
  (void)in_sizes; (void)n_in; (void)out_size; (void)ws_size;
  // ----- input leaves (jax pytree order: sorted dict keys) -----
  const float* x      = (const float*)d_in[0];
  const int*   y_type = (const int*)d_in[1];
  const float* y_rest = (const float*)d_in[2];
  const float* nerr   = (const float*)d_in[3];
  const int*   ei     = (const int*)d_in[4];   // [2, E]
  const float* eattr  = (const float*)d_in[5]; // [E, 5]
  const int*   dnode  = (const int*)d_in[6];   // [E, 2]
  // params: edge_in(b0,b1,b2,beta,g,w0,w1,w2)=7..14, embed=15,
  // mp[i]: base 16+16*i (edge 8 leaves, node 8 leaves), node_in=80..87,
  // node_out(b0,b1,b2,w0,w1,w2)=88..93
  const int EI = 7, EMB = 15, MP0 = 16, NI = 80, NO = 88;
  auto F = [&](int i) { return (const float*)d_in[i]; };

  // ----- workspace layout -----
  size_t off = 0;
  auto alloc = [&](size_t bytes) -> char* {
    off = (off + 255) & ~(size_t)255;
    char* p = (char*)d_ws + off;
    off += bytes;
    return p;
  };
  _Float16* h16  = (_Float16*)alloc((size_t)NN * H * 2);
  _Float16* e16  = (_Float16*)alloc((size_t)NE * H * 2);
  float*    aggr = (float*)alloc((size_t)NN * H * 4);
  int*      effi = (int*)alloc((size_t)NE * 4);
  int*      effj = (int*)alloc((size_t)NE * 4);
  // nf/ea alias the aggr region (only live before the MP loop)
  _Float16* nf16 = (_Float16*)aggr;
  _Float16* ea16 = (_Float16*)((char*)aggr + (16u << 20));

  // packed weights (f16 B-fragment layout)
  _Float16 *niw[3], *eiw[3], *mew[4][3], *mnw[4][3], *now[2];
  for (int l = 0; l < 3; ++l) niw[l] = (_Float16*)alloc((l ? 4 : 1) * 8192);
  for (int l = 0; l < 3; ++l) eiw[l] = (_Float16*)alloc((l ? 4 : 1) * 8192);
  for (int i = 0; i < 4; ++i) {
    for (int l = 0; l < 3; ++l) mew[i][l] = (_Float16*)alloc((l ? 4 : 12) * 8192);
    for (int l = 0; l < 3; ++l) mnw[i][l] = (_Float16*)alloc((l ? 4 : 8) * 8192);
  }
  for (int l = 0; l < 2; ++l) now[l] = (_Float16*)alloc(4 * 8192);

  // ----- pack all weight matrices -----
  struct PJ { int leaf; int kin; int ks; _Float16* dst; };
  PJ jobs[32]; int nj = 0;
  jobs[nj++] = {NI + 5, 12, 1, niw[0]};
  jobs[nj++] = {NI + 6, 128, 4, niw[1]};
  jobs[nj++] = {NI + 7, 128, 4, niw[2]};
  jobs[nj++] = {EI + 5, 5, 1, eiw[0]};
  jobs[nj++] = {EI + 6, 128, 4, eiw[1]};
  jobs[nj++] = {EI + 7, 128, 4, eiw[2]};
  for (int i = 0; i < 4; ++i) {
    int b = MP0 + 16 * i;
    jobs[nj++] = {b + 5, 384, 12, mew[i][0]};
    jobs[nj++] = {b + 6, 128, 4, mew[i][1]};
    jobs[nj++] = {b + 7, 128, 4, mew[i][2]};
    jobs[nj++] = {b + 13, 256, 8, mnw[i][0]};
    jobs[nj++] = {b + 14, 128, 4, mnw[i][1]};
    jobs[nj++] = {b + 15, 128, 4, mnw[i][2]};
  }
  jobs[nj++] = {NO + 3, 128, 4, now[0]};
  jobs[nj++] = {NO + 4, 128, 4, now[1]};
  for (int j = 0; j < nj; ++j) {
    int total = jobs[j].ks * 4096;
    pack_w_kernel<<<(total + 255) / 256, 256, 0, stream>>>(
        F(jobs[j].leaf), jobs[j].dst, jobs[j].kin, total);
  }

  // ----- build encoder inputs + effective gather indices -----
  build_nf_kernel<<<NN / 256, 256, 0, stream>>>(x, y_type, y_rest, nerr,
                                                F(EMB), nf16);
  build_ea_kernel<<<NE / 256, 256, 0, stream>>>(eattr, ea16);
  eff_idx_kernel<<<NE / 256, 256, 0, stream>>>(ei, dnode, effi, effj);

  // ----- encoders -----
  mlp_enc_kernel<<<NN / 256, 256, 0, stream>>>(
      nf16, niw[0], niw[1], niw[2], F(NI + 0), F(NI + 1), F(NI + 2),
      F(NI + 4), F(NI + 3), h16);
  mlp_enc_kernel<<<NE / 256, 256, 0, stream>>>(
      ea16, eiw[0], eiw[1], eiw[2], F(EI + 0), F(EI + 1), F(EI + 2),
      F(EI + 4), F(EI + 3), e16);

  // ----- message-passing loop -----
  const int* dst = ei + NE;
  for (int i = 0; i < 4; ++i) {
    int b = MP0 + 16 * i;
    zero_aggr_kernel<<<(NN * H / 4 + 255) / 256, 256, 0, stream>>>(aggr);
    edge_mp_kernel<<<NE / 256, 256, 0, stream>>>(
        h16, e16, effi, effj, dst, mew[i][0], mew[i][1], mew[i][2],
        F(b + 0), F(b + 1), F(b + 2), F(b + 4), F(b + 3), aggr);
    node_mp_kernel<<<NN / 256, 256, 0, stream>>>(
        h16, aggr, mnw[i][0], mnw[i][1], mnw[i][2],
        F(b + 8), F(b + 9), F(b + 10), F(b + 12), F(b + 11));
  }

  // ----- decoder + second tuple output -----
  decode_kernel<<<NN / 256, 256, 0, stream>>>(
      h16, now[0], now[1], F(NO + 0), F(NO + 1), F(NO + 3 + 2), F(NO + 2),
      (float*)d_out);
  copy_err_kernel<<<NN / 256, 256, 0, stream>>>(nerr, (float*)d_out + NN);
}
